// SketchGCN_26551487824432
// MI455X (gfx1250) — matrix operand order repaired and checked by reading, not compiled, not run on hardware
//
#include <hip/hip_runtime.h>
#include <hip/hip_bf16.h>

typedef __attribute__((ext_vector_type(8)))  __bf16 v8bf;
typedef __attribute__((ext_vector_type(16))) __bf16 v16bf;
typedef __attribute__((ext_vector_type(8)))  float  v8f;
typedef __attribute__((ext_vector_type(4)))  float  v4f;

#define N_ROWS 8192
#define K_DIM  8192

// ---------------------------------------------------------------------------
// fp32 -> bf16 conversion of A (done once, reused by 3 GEMMs).
// fp32 input is dead after this -> non-temporal loads; bf16 output (128 MB)
// fits the 192 MB L2 -> regular-temporal stores so later GEMMs can hit L2.
// ---------------------------------------------------------------------------
__global__ __launch_bounds__(256) void cvt_bf16_kernel(const float* __restrict__ in,
                                                       __bf16* __restrict__ out,
                                                       long long n) {
  long long i = ((long long)blockIdx.x * 256 + threadIdx.x) * 8;
  if (i + 8 > n) return;
  v4f x = __builtin_nontemporal_load((const v4f*)(in + i));
  v4f y = __builtin_nontemporal_load((const v4f*)(in + i + 4));
  union { v8bf v; __bf16 e[8]; } o;
  o.e[0] = (__bf16)x.x; o.e[1] = (__bf16)x.y; o.e[2] = (__bf16)x.z; o.e[3] = (__bf16)x.w;
  o.e[4] = (__bf16)y.x; o.e[5] = (__bf16)y.y; o.e[6] = (__bf16)y.z; o.e[7] = (__bf16)y.w;
  *(v8bf*)(out + i) = o.v;
}

// ---------------------------------------------------------------------------
// Small projection: YT[c][i] = (c < COUT) ? sum_j X[i][j]*W[j][c] + b[c] : 0
// Output TRANSPOSED bf16 so gemmA's B fragments are contiguous 32B slices.
// ---------------------------------------------------------------------------
template<int CIN, int COUT, int CPAD>
__global__ __launch_bounds__(256) void rowmm_kernel(const float* __restrict__ X,
                                                    const float* __restrict__ W,
                                                    const float* __restrict__ bias,
                                                    __bf16* __restrict__ YT) {
  __shared__ float Ws[CIN * COUT];
  __shared__ float bs[COUT];
  const int t = threadIdx.x;
  for (int idx = t; idx < CIN * COUT; idx += 256) Ws[idx] = W[idx];
  if (t < COUT) bs[t] = bias[t];
  __syncthreads();

  const int i = blockIdx.x * 256 + t;
  const float* xr = X + (size_t)i * CIN;
  float acc[COUT];
#pragma unroll
  for (int c = 0; c < COUT; ++c) acc[c] = bs[c];
  for (int j = 0; j < CIN; ++j) {
    const float xv = xr[j];
#pragma unroll
    for (int c = 0; c < COUT; ++c) acc[c] = fmaf(xv, Ws[j * COUT + c], acc[c]);
  }
#pragma unroll
  for (int c = 0; c < CPAD; ++c)
    YT[(size_t)c * N_ROWS + i] = (__bf16)((c < COUT) ? acc[c] : 0.0f);
}

// ---------------------------------------------------------------------------
// Z = A(bf16) @ Y with fused cubic polynomial (+ optional residual), fp32 out.
// 4 waves / 128 threads per WG, 16 rows per wave -> 64 rows/WG, 128 WGs.
// Manually 2x-unrolled, double-buffered, software-pipelined main loop:
// explicit ping-pong registers (no a_cur=a_nxt copies), distinct B-fragment
// registers (all 8 ds_load_b128 in flight), one barrier per k-step.
// ---------------------------------------------------------------------------
#define GA_BLOCK 128
#define GA_MWG   64
#define BS_LD    48   // padded LDS row: 96B rows keep 32B-aligned fragments

template<int NT, bool RES>
__global__ __launch_bounds__(GA_BLOCK) void gemmA_kernel(
    const __bf16* __restrict__ Abf,   // [8192, 8192] row-major bf16
    const __bf16* __restrict__ YT,    // [NT*16, 8192] row-major bf16 (B^T)
    const float*  __restrict__ res,   // residual [8192, 64] or nullptr
    const float*  __restrict__ coef,  // 3 polynomial coefficients
    float*        __restrict__ out,   // [8192, ldc]
    int ldc, int nout) {
  constexpr int CPAD = NT * 16;
  constexpr int STG  = CPAD * 4;                    // b128 staging elements/block
  __shared__ __align__(32) __bf16 Bs[2][CPAD][BS_LD];

  const int t = threadIdx.x;
  __builtin_assume(t < GA_BLOCK);                   // folds do1 for NT==4
  const int lane = t & 31;
  const int wave = t >> 5;
  const int m0   = blockIdx.x * GA_MWG + wave * 16;

  const float c0 = coef[0], c1 = coef[1], c2 = coef[2];  // early scalar loads

  // A fragment: lane<16 covers K {0..15}, lane>=16 covers {16..31}
  const __bf16* Arow = Abf + (size_t)(m0 + (lane & 15)) * K_DIM + ((lane >> 4) * 16);
  // B fragment: column N = 16n + (lane&15); K half selected by lane>>4
  const int boff = (lane & 15) * BS_LD + (lane >> 4) * 16;

  // Fixed staging assignment: exactly 2 b128s per thread (2nd guarded, NT==3)
  const int e1   = t + GA_BLOCK;
  const int c0s  = t >> 2,  k0s = (t & 3) * 8;
  const int c1s  = e1 >> 2, k1s = (e1 & 3) * 8;
  const bool do1 = (e1 < STG);                      // constant true for NT==4
  const __bf16* yp0 = YT + (size_t)c0s * K_DIM + k0s;
  const __bf16* yp1 = YT + (size_t)c1s * K_DIM + k1s;

  union AF { v16bf v; v8bf h[2]; };
  v8f acc[NT] = {};

  auto stage_load = [&](int kblk, v8bf& g0, v8bf& g1) {
    g0 = *(const v8bf*)(yp0 + kblk);
    if (do1) g1 = *(const v8bf*)(yp1 + kblk);
  };
  auto stage_store = [&](int buf, v8bf g0, v8bf g1) {
    *(v8bf*)&Bs[buf][c0s][k0s] = g0;
    if (do1) *(v8bf*)&Bs[buf][c1s][k1s] = g1;
  };
  auto load_a = [&](int kblk, AF& a) {
    a.h[0] = *(const v8bf*)(Arow + kblk);
    a.h[1] = *(const v8bf*)(Arow + kblk + 8);
  };
  auto compute = [&](int buf, const AF& a) {
    const __bf16* bb = &Bs[buf][0][0] + boff;
    v16bf bf[NT];                                    // distinct regs per tile
#pragma unroll
    for (int n = 0; n < NT; ++n) bf[n] = *(const v16bf*)(bb + n * 16 * BS_LD);
#pragma unroll
    for (int n = 0; n < NT; ++n)
      acc[n] = __builtin_amdgcn_wmma_f32_16x16x32_bf16(
          false, a.v, false, bf[n], (short)0, acc[n], false, false);
  };

  // --- prologue: stage k-block 0 into buffer 0, preload A fragment ---
  AF a0, a1;
  v8bf g0, g1;
  stage_load(0, g0, g1);
  stage_store(0, g0, g1);
  load_a(0, a0);
  __syncthreads();

  // --- main loop: two k-steps (64 K) per iteration, explicit ping-pong ---
  for (int k0 = 0; k0 < K_DIM; k0 += 64) {
    // step A: prefetch k0+32 -> (g, a1); compute buf0/a0; store -> buf1
    stage_load(k0 + 32, g0, g1);
    load_a(k0 + 32, a1);
    compute(0, a0);
    stage_store(1, g0, g1);
    __syncthreads();

    // step B: prefetch k0+64 -> (g, a0) [guarded]; compute buf1/a1; -> buf0
    const bool more = (k0 + 64) < K_DIM;
    if (more) {
      stage_load(k0 + 64, g0, g1);
      load_a(k0 + 64, a0);
    }
    compute(1, a1);
    if (more) stage_store(0, g0, g1);
    __syncthreads();
  }

  // --- epilogue: poly (c0*z + c1*z^2 + c2*z^3), residual, masked store ---
  const int col0 = lane & 15;
  const int rsel = (lane >> 4) * 8;   // C/D layout: upper lanes hold M = r+8
#pragma unroll
  for (int n = 0; n < NT; ++n) {
    const int col = n * 16 + col0;
#pragma unroll
    for (int r = 0; r < 8; ++r) {
      const int row = m0 + r + rsel;
      const float z = acc[n][r];
      const float pz = z * z;
      float v = c0 * z + c1 * pz + c2 * (pz * z);
      if (RES) v += res[(size_t)row * 64 + col];
      if (col < nout) out[(size_t)row * ldc + col] = v;
    }
  }
}

// ---------------------------------------------------------------------------
extern "C" void kernel_launch(void* const* d_in, const int* in_sizes, int n_in,
                              void* d_out, int out_size, void* d_ws, size_t ws_size,
                              hipStream_t stream) {
  (void)in_sizes; (void)n_in; (void)out_size; (void)ws_size;
  const float* X  = (const float*)d_in[0];   // nf_mats   [8192,128]
  const float* A  = (const float*)d_in[1];   // conv_mats [8192,8192]
  const float* W1 = (const float*)d_in[2];
  const float* b1 = (const float*)d_in[3];
  const float* c1 = (const float*)d_in[4];
  const float* W2 = (const float*)d_in[5];
  const float* b2 = (const float*)d_in[6];
  const float* c2 = (const float*)d_in[7];
  const float* W3 = (const float*)d_in[8];
  const float* b3 = (const float*)d_in[9];
  const float* c3 = (const float*)d_in[10];
  float* out = (float*)d_out;

  char* w = (char*)d_ws;
  const size_t A_BYTES = (size_t)N_ROWS * K_DIM * 2;        // 128 MB
  __bf16* Abf = (__bf16*)w;  w += A_BYTES;
  __bf16* YT  = (__bf16*)w;  w += (size_t)64 * N_ROWS * 2;  // 1 MB (covers CPAD<=64)
  float*  x1  = (float*)w;   w += (size_t)N_ROWS * 64 * 4;  // 2 MB
  float*  x2  = (float*)w;

  const long long nA = (long long)N_ROWS * K_DIM;
  cvt_bf16_kernel<<<(int)(nA / (256 * 8)), 256, 0, stream>>>(A, Abf, nA);

  // Layer 1
  rowmm_kernel<128, 64, 64><<<N_ROWS / 256, 256, 0, stream>>>(X, W1, b1, YT);
  gemmA_kernel<4, false><<<N_ROWS / GA_MWG, GA_BLOCK, 0, stream>>>(
      Abf, YT, nullptr, c1, x1, 64, 64);
  // Layer 2 (residual)
  rowmm_kernel<64, 64, 64><<<N_ROWS / 256, 256, 0, stream>>>(x1, W2, b2, YT);
  gemmA_kernel<4, true><<<N_ROWS / GA_MWG, GA_BLOCK, 0, stream>>>(
      Abf, YT, x1, c2, x2, 64, 64);
  // Layer 3 (40 cols padded to 48 = 3 N-tiles)
  rowmm_kernel<64, 40, 48><<<N_ROWS / 256, 256, 0, stream>>>(x2, W3, b3, YT);
  gemmA_kernel<3, false><<<N_ROWS / GA_MWG, GA_BLOCK, 0, stream>>>(
      Abf, YT, nullptr, c3, out, 40, 40);
}
